// MultiHeadAttention_70815420776821
// MI455X (gfx1250) — matrix operand order repaired and checked by reading, not compiled, not run on hardware
//
#include <hip/hip_runtime.h>
#include <cstdint>

typedef __bf16 bf16;
typedef __attribute__((ext_vector_type(16))) __bf16 v16bf;
typedef __attribute__((ext_vector_type(8)))  __bf16 v8bf;
typedef __attribute__((ext_vector_type(8)))  float  v8f;

#define DEV static __device__ __forceinline__

constexpr int Bq = 4, Sq = 2048, Dm = 1024, Hh = 16, HDd = 64;

// round-to-nearest-even f32 -> bf16 bits
DEV uint16_t f2bf_bits(float f) {
  uint32_t u = __builtin_bit_cast(uint32_t, f);
  u += 0x7FFFu + ((u >> 16) & 1u);
  return (uint16_t)(u >> 16);
}
DEV bf16 f2bf(float f) { return __builtin_bit_cast(bf16, f2bf_bits(f)); }

// A-matrix fragment (16x32 bf16): lane half h reads K = {h*8+0..7, 16+h*8+0..7} of its row
DEV v16bf load_afrag(const bf16* row, int h) {
  v8bf lo = *(const v8bf*)(row + h * 8);
  v8bf hi = *(const v8bf*)(row + 16 + h * 8);
  return __builtin_shufflevector(lo, hi, 0,1,2,3,4,5,6,7,8,9,10,11,12,13,14,15);
}
// B-matrix fragment (32x16 bf16): lane half h reads K = {h*8+0..7, 16+h*8+0..7} of its column
DEV v16bf load_bfrag(const bf16* row, int h) {
  v8bf lo = *(const v8bf*)(row + h * 8);
  v8bf hi = *(const v8bf*)(row + 16 + h * 8);
  return __builtin_shufflevector(lo, hi, 0,1,2,3,4,5,6,7,8,9,10,11,12,13,14,15);
}

DEV v8f wmma_bf16(v16bf a, v16bf b, v8f c) {
  return __builtin_amdgcn_wmma_f32_16x16x32_bf16(false, a, false, b, (short)0, c, false, false);
}

// ---- CDNA5 async global->LDS (ASYNCcnt path, no VGPR staging) --------------
// LDS operand is wave-relative byte offset == low 32 bits of generic pointer
// (flat aperture: LDS address is the low 32 address bits).
DEV void async_ld_b128(const void* lds_dst, const void* gsrc) {
  unsigned l = (unsigned)(uintptr_t)lds_dst;
  asm volatile("global_load_async_to_lds_b128 %0, %1, off"
               :: "v"(l), "v"((uint64_t)(uintptr_t)gsrc)
               : "memory");
}
DEV void wait_async0() { asm volatile("s_wait_asynccnt 0x0" ::: "memory"); }

// ---- cross-lane reductions within 16-lane groups via ds_swizzle (xor mode) -
template <int IMM> DEV float swz_max(float v) {
  int o = __builtin_amdgcn_ds_swizzle(__builtin_bit_cast(int, v), IMM);
  return fmaxf(v, __builtin_bit_cast(float, o));
}
template <int IMM> DEV float swz_add(float v) {
  int o = __builtin_amdgcn_ds_swizzle(__builtin_bit_cast(int, v), IMM);
  return v + __builtin_bit_cast(float, o);
}
// xor-swizzle imm: offset[15]=0, and_mask=0x1f, or_mask=0, xor_mask=m<<10
DEV float red_max16(float v) {
  v = swz_max<0x041f>(v); v = swz_max<0x081f>(v);
  v = swz_max<0x101f>(v); v = swz_max<0x201f>(v);
  return v;
}
DEV float red_sum16(float v) {
  v = swz_add<0x041f>(v); v = swz_add<0x081f>(v);
  v = swz_add<0x101f>(v); v = swz_add<0x201f>(v);
  return v;
}

// ---------------------------------------------------------------------------
// Generic GEMM: Out[M,N] = A[M,K] @ W[N,K]^T + bias, M=8192, N=K=1024
// Block tile 128x128, BK=32, 8 waves (4x2), each wave: 32x64 = 2x4 WMMA tiles.
// EPI 0: bf16 out, [B,H,S,HD] (Q,K).  EPI 1: bf16 out, [B,H,HD,S] (V^T).
// EPI 2: fp32 out, [M,N] (final projection).  ABF16: A is bf16 (else fp32).
// ---------------------------------------------------------------------------
template <int EPI, bool ABF16>
__global__ __launch_bounds__(256) void gemm_bias_kernel(
    const void* __restrict__ Av, const float* __restrict__ W,
    const float* __restrict__ bias, void* __restrict__ Outv) {
  constexpr int K = Dm, BK = 32, NK = K / BK;
  __shared__ bf16 As[2][128][40];  // 32 + 8 pad (80B stride)
  __shared__ bf16 Ws[2][128][40];

  const int t = threadIdx.x;
  const int lane = t & 31;
  const int wave = t >> 5;
  const int h = (lane >> 4) & 1;
  const int lm = lane & 15;
  const int wm = wave >> 1, wn = wave & 1;
  const int n0 = blockIdx.x * 128;
  const int m0 = blockIdx.y * 128;

  const float* Af = (const float*)Av;
  const bf16*  Ab = (const bf16*)Av;

  v8f acc[2][4];
#pragma unroll
  for (int i = 0; i < 2; i++)
#pragma unroll
    for (int j = 0; j < 4; j++)
#pragma unroll
      for (int r = 0; r < 8; r++) acc[i][j][r] = 0.0f;

  float4 ra[4]; uint4 rab[2]; float4 rw[4];

  auto gload = [&](int k0) {
    if constexpr (ABF16) {
#pragma unroll
      for (int i = 0; i < 2; i++) {
        int idx = t + i * 256, row = idx >> 2, c = (idx & 3) * 8;
        rab[i] = *(const uint4*)(Ab + (size_t)(m0 + row) * K + k0 + c);
      }
    } else {
#pragma unroll
      for (int i = 0; i < 4; i++) {
        int idx = t + i * 256, row = idx >> 3, c = (idx & 7) * 4;
        ra[i] = *(const float4*)(Af + (size_t)(m0 + row) * K + k0 + c);
      }
    }
#pragma unroll
    for (int i = 0; i < 4; i++) {
      int idx = t + i * 256, row = idx >> 3, c = (idx & 7) * 4;
      rw[i] = *(const float4*)(W + (size_t)(n0 + row) * K + k0 + c);
    }
    __builtin_prefetch(W + (size_t)(n0 + (t >> 3)) * K + k0 + BK, 0, 1);
  };
  auto sstore = [&](int buf) {
    if constexpr (ABF16) {
#pragma unroll
      for (int i = 0; i < 2; i++) {
        int idx = t + i * 256, row = idx >> 2, c = (idx & 3) * 8;
        *(uint4*)(&As[buf][row][c]) = rab[i];
      }
    } else {
#pragma unroll
      for (int i = 0; i < 4; i++) {
        int idx = t + i * 256, row = idx >> 3, c = (idx & 7) * 4;
        uint2 p;
        p.x = (uint32_t)f2bf_bits(ra[i].x) | ((uint32_t)f2bf_bits(ra[i].y) << 16);
        p.y = (uint32_t)f2bf_bits(ra[i].z) | ((uint32_t)f2bf_bits(ra[i].w) << 16);
        *(uint2*)(&As[buf][row][c]) = p;
      }
    }
#pragma unroll
    for (int i = 0; i < 4; i++) {
      int idx = t + i * 256, row = idx >> 3, c = (idx & 7) * 4;
      uint2 p;
      p.x = (uint32_t)f2bf_bits(rw[i].x) | ((uint32_t)f2bf_bits(rw[i].y) << 16);
      p.y = (uint32_t)f2bf_bits(rw[i].z) | ((uint32_t)f2bf_bits(rw[i].w) << 16);
      *(uint2*)(&Ws[buf][row][c]) = p;
    }
  };

  gload(0);
  sstore(0);
  __syncthreads();

  for (int kt = 0; kt < NK; ++kt) {
    const int cur = kt & 1;
    if (kt + 1 < NK) gload((kt + 1) * BK);

    v16bf afr[2], bfr[4];
#pragma unroll
    for (int mt = 0; mt < 2; mt++)
      afr[mt] = load_afrag(&As[cur][wm * 32 + mt * 16 + lm][0], h);
#pragma unroll
    for (int nt = 0; nt < 4; nt++)
      bfr[nt] = load_bfrag(&Ws[cur][wn * 64 + nt * 16 + lm][0], h);
#pragma unroll
    for (int mt = 0; mt < 2; mt++)
#pragma unroll
      for (int nt = 0; nt < 4; nt++)
        acc[mt][nt] = wmma_bf16(afr[mt], bfr[nt], acc[mt][nt]);

    if (kt + 1 < NK) sstore(cur ^ 1);
    __syncthreads();
  }

  // epilogue: C frag element [r] = row (r + 8*h), col (lane&15)
#pragma unroll
  for (int mt = 0; mt < 2; mt++) {
#pragma unroll
    for (int nt = 0; nt < 4; nt++) {
      const int gn = n0 + wn * 64 + nt * 16 + lm;
      const float bv = bias[gn];
#pragma unroll
      for (int r = 0; r < 8; r++) {
        const int gm = m0 + wm * 32 + mt * 16 + r + 8 * h;
        const float v = acc[mt][nt][r] + bv;
        if constexpr (EPI == 2) {
          ((float*)Outv)[(size_t)gm * Dm + gn] = v;
        } else {
          const int b = gm / Sq, s = gm % Sq;
          const int hh = gn / HDd, hd = gn % HDd;
          size_t off;
          if constexpr (EPI == 0)
            off = (((size_t)(b * Hh + hh)) * Sq + s) * HDd + hd;
          else
            off = (((size_t)(b * Hh + hh)) * HDd + hd) * Sq + s;
          ((bf16*)Outv)[off] = f2bf(v);
        }
      }
    }
  }
}

// ---------------------------------------------------------------------------
// Flash attention: grid (S/128, B*H), 8 waves; each wave owns a 16-row q-tile,
// streams 64-key blocks. K/V tiles move global->LDS via async DMA (ASYNCcnt),
// double-buffered; online softmax in registers with ds_swizzle reductions;
// P transposed C->A layout via per-wave LDS.
// ---------------------------------------------------------------------------
__global__ __launch_bounds__(256) void flash_attn_kernel(
    const bf16* __restrict__ Qb, const bf16* __restrict__ Kb,
    const bf16* __restrict__ Vt, bf16* __restrict__ Attn) {
  __shared__ bf16 Ks[2][64][72];   // 64 keys x 64 d   (144B stride)
  __shared__ bf16 Vs[2][64][72];   // 64 hd   x 64 keys
  __shared__ bf16 Ps[8][16][72];   // per-wave P tile 16 q x 64 keys

  const int t = threadIdx.x;
  const int lane = t & 31;
  const int wave = t >> 5;
  const int h = (lane >> 4) & 1;
  const int lm = lane & 15;
  const int bh = blockIdx.y;                  // b*H + h
  const int q0 = blockIdx.x * 128 + wave * 16;

  // per-wave Q fragments (16 x 64), loaded once from global
  const bf16* qrow = Qb + ((size_t)bh * Sq + q0 + lm) * HDd;
  v16bf qf[2];
#pragma unroll
  for (int ks = 0; ks < 2; ++ks) qf[ks] = load_afrag(qrow + ks * 32, h);

  v8f of[4];
  float mrun[8], lrun[8];
#pragma unroll
  for (int nt = 0; nt < 4; nt++)
#pragma unroll
    for (int r = 0; r < 8; r++) of[nt][r] = 0.0f;
#pragma unroll
  for (int r = 0; r < 8; r++) { mrun[r] = -1e30f; lrun[r] = 0.0f; }

  // async global->LDS issue for one 64-key block (4x b128 per thread)
  auto issue = [&](int buf, int kblk) {
#pragma unroll
    for (int i = 0; i < 2; i++) {
      int idx = t + i * 256, row = idx >> 3, c = (idx & 7) * 8;
      async_ld_b128(&Ks[buf][row][c],
                    Kb + ((size_t)bh * Sq + kblk + row) * HDd + c);
      async_ld_b128(&Vs[buf][row][c],
                    Vt + ((size_t)bh * HDd + row) * Sq + kblk + c);
    }
  };

  issue(0, 0);
  wait_async0();
  __syncthreads();

  constexpr int NBLK = Sq / 64;
  for (int blk = 0; blk < NBLK; ++blk) {
    const int cur = blk & 1;
    if (blk + 1 < NBLK) issue(cur ^ 1, (blk + 1) * 64);

    // scores S = Q K^T (16 x 64)
    v8f sc[4];
#pragma unroll
    for (int nt = 0; nt < 4; nt++) {
#pragma unroll
      for (int r = 0; r < 8; r++) sc[nt][r] = 0.0f;
#pragma unroll
      for (int ks = 0; ks < 2; ++ks) {
        v16bf kf = load_bfrag(&Ks[cur][nt * 16 + lm][ks * 32], h);
        sc[nt] = wmma_bf16(qf[ks], kf, sc[nt]);
      }
    }

    // online softmax per query row (row r+8*h -> element r, 16-lane group)
#pragma unroll
    for (int r = 0; r < 8; r++) {
      float mx = -1e30f;
#pragma unroll
      for (int nt = 0; nt < 4; nt++) {
        sc[nt][r] *= 0.125f;                  // 1/sqrt(64)
        mx = fmaxf(mx, sc[nt][r]);
      }
      mx = red_max16(mx);
      const float mnew = fmaxf(mrun[r], mx);
      const float corr = __expf(mrun[r] - mnew);
      mrun[r] = mnew;
      float rs = 0.0f;
#pragma unroll
      for (int nt = 0; nt < 4; nt++) {
        const float p = __expf(sc[nt][r] - mnew);
        sc[nt][r] = p;
        rs += p;
      }
      rs = red_sum16(rs);
      lrun[r] = lrun[r] * corr + rs;
#pragma unroll
      for (int nt = 0; nt < 4; nt++) of[nt][r] *= corr;
    }

    // P: C layout -> A layout through per-wave LDS (wave-internal DS ordering)
#pragma unroll
    for (int nt = 0; nt < 4; nt++)
#pragma unroll
      for (int r = 0; r < 8; r++)
        Ps[wave][r + 8 * h][nt * 16 + lm] = f2bf(sc[nt][r]);

    // O += P V  (K-dim = 64 keys -> 2 k-steps, 4 hd n-tiles)
#pragma unroll
    for (int ks = 0; ks < 2; ++ks) {
      v16bf pf = load_afrag(&Ps[wave][lm][ks * 32], h);
#pragma unroll
      for (int nt = 0; nt < 4; nt++) {
        v16bf vf = load_bfrag(&Vs[cur][nt * 16 + lm][ks * 32], h);
        of[nt] = wmma_bf16(pf, vf, of[nt]);
      }
    }

    wait_async0();        // next buffer's DMA landed
    __syncthreads();
  }

  // normalize and write attention output as bf16 [B,S,D]
  const int b = bh / Hh, hh = bh % Hh;
#pragma unroll
  for (int nt = 0; nt < 4; nt++)
#pragma unroll
    for (int r = 0; r < 8; r++) {
      const float v = of[nt][r] / lrun[r];
      const size_t off =
          ((size_t)b * Sq + q0 + r + 8 * h) * Dm + hh * HDd + nt * 16 + lm;
      Attn[off] = f2bf(v);
    }
}

// ---------------------------------------------------------------------------
extern "C" void kernel_launch(void* const* d_in, const int* in_sizes, int n_in,
                              void* d_out, int out_size, void* d_ws, size_t ws_size,
                              hipStream_t stream) {
  (void)in_sizes; (void)n_in; (void)out_size; (void)ws_size;
  const float* query = (const float*)d_in[0];
  const float* key_  = (const float*)d_in[1];
  const float* value = (const float*)d_in[2];
  const float* Wq = (const float*)d_in[3];
  const float* bq = (const float*)d_in[4];
  const float* Wk = (const float*)d_in[5];
  const float* bk = (const float*)d_in[6];
  const float* Wv = (const float*)d_in[7];
  const float* bv = (const float*)d_in[8];
  const float* Wo = (const float*)d_in[9];
  const float* bo = (const float*)d_in[10];

  char* ws = (char*)d_ws;
  bf16* Qb   = (bf16*)(ws);                          // [B,H,S,64]  16 MiB
  bf16* Kb   = (bf16*)(ws + ((size_t)16 << 20));     // [B,H,S,64]  16 MiB
  bf16* Vt   = (bf16*)(ws + ((size_t)32 << 20));     // [B,H,64,S]  16 MiB
  bf16* Attn = (bf16*)(ws + ((size_t)48 << 20));     // [B,S,D]     16 MiB

  const dim3 blk(256);
  const dim3 gg(Dm / 128, (Bq * Sq) / 128);          // (8, 64)
  const dim3 gf(Sq / 128, Bq * Hh);                  // (16, 64)

  hipLaunchKernelGGL((gemm_bias_kernel<0, false>), gg, blk, 0, stream,
                     (const void*)query, Wq, bq, (void*)Qb);
  hipLaunchKernelGGL((gemm_bias_kernel<0, false>), gg, blk, 0, stream,
                     (const void*)key_, Wk, bk, (void*)Kb);
  hipLaunchKernelGGL((gemm_bias_kernel<1, false>), gg, blk, 0, stream,
                     (const void*)value, Wv, bv, (void*)Vt);
  hipLaunchKernelGGL(flash_attn_kernel, gf, blk, 0, stream, Qb, Kb, Vt, Attn);
  hipLaunchKernelGGL((gemm_bias_kernel<2, true>), gg, blk, 0, stream,
                     (const void*)Attn, Wo, bo, d_out);
}
